// ExpRNN_12713103196529
// MI455X (gfx1250) — matrix operand order repaired and checked by reading, not compiled, not run on hardware
//
#include <hip/hip_runtime.h>
#include <hip/hip_bf16.h>
#include <stdint.h>

// ExpRNN (modReLU 2-layer RNN), B=64 T=512 D=H=1024.
// Persistent WMMA kernel: 32 blocks x 128 threads = 128 wave32s; each wave owns
// a 16x32 tile of the [64,1024] hidden-state GEMM. bf16 operands; each block's
// 256 KB weight slice (32 cols x 1024 K x 4 matrices) is staged once into the
// WGP's 320 KB LDS and served by ds_load_b128 for all 512 timesteps, so the
// steady-state inner loop is A-loads (global, L2-hot) + B-loads (LDS) + WMMA.

#define B_   64
#define T_   512
#define D_   1024
#define H_   1024
#define BH_  (B_ * H_)
#define NBLOCKS 32
#define NTHREADS 128
#define SLICE_ (32 * 1024)          // elements per weight slice in LDS

typedef __attribute__((ext_vector_type(16))) __bf16 v16bf;
typedef __attribute__((ext_vector_type(8)))  float  v8f;

union BFrag {
    v16bf v;
    uint4 q[2];
    unsigned short u[16];
};

__device__ inline unsigned short f2bf(float f) {
    unsigned u = __float_as_uint(f);
    unsigned r = 0x7FFFu + ((u >> 16) & 1u);     // round-to-nearest-even
    return (unsigned short)((u + r) >> 16);
}

// A-matrix 16x32 bf16 fragment, row-major bf16 source, row stride 1024.
// ISA layout (16-bit A 16x32): lane<16 -> row M=lane, K = {0..7, 16..23};
// lane>=16 -> row M=lane-16, K = {8..15, 24..31}. Two 16B loads per lane.
__device__ inline v16bf load_a(const unsigned short* __restrict__ base,
                               int row, int kb, int lane) {
    const int kL = (lane >> 4) * 8;
    const unsigned short* p = base + (size_t)row * 1024 + kb + kL;
    BFrag a;
    a.q[0] = *(const uint4*)(p);        // K = kL .. kL+7
    a.q[1] = *(const uint4*)(p + 16);   // K = 16+kL .. 16+kL+7
    return a.v;
}

// B-matrix 32x16 bf16 fragment from an LDS weight slice [32 cols][1024 K].
// ISA layout (16-bit B): lane<16 -> col N=lane, K = kb+{0..7, 16..23};
// lane>=16 -> col N=lane-16, K = kb+{8..15, 24..31}. Two 16B loads per lane.
__device__ inline v16bf load_b_lds(const unsigned short* Ws,
                                   int colLocal, int kb, int lane) {
    const int khi = (lane >> 4) * 8;
    const uint4* p = (const uint4*)(Ws + colLocal * 1024 + kb + khi);
    BFrag b;
    b.q[0] = p[0];
    b.q[1] = p[2];
    return b.v;
}

// Accumulate a 16x32 tile of (A[bf16, global] @ W^T[bf16, LDS slice]) over K=1024.
__device__ inline void gemm_tile(v8f& c0, v8f& c1,
                                 const unsigned short* __restrict__ A,
                                 const unsigned short* Ws,   // LDS, 32x1024
                                 int mTile, int lane) {
    const int rowA = mTile * 16 + (lane & 15);
    const int col0 = lane & 15;
    const int col1 = col0 + 16;
#pragma unroll 4
    for (int kb = 0; kb < 1024; kb += 32) {
        v16bf a  = load_a(A, rowA, kb, lane);
        v16bf b0 = load_b_lds(Ws, col0, kb, lane);
        v16bf b1 = load_b_lds(Ws, col1, kb, lane);
        c0 = __builtin_amdgcn_wmma_f32_16x16x32_bf16(false, a, false, b0, (short)0, c0, false, false);
        c1 = __builtin_amdgcn_wmma_f32_16x16x32_bf16(false, a, false, b1, (short)0, c1, false, false);
    }
}

// C/D layout: lane<16 -> N=lane, M=vgpr i; lane>=16 -> N=lane-16, M=8+i.
// Writes bf16 hidden state (always) + optional f32 outputs. Biases in registers.
__device__ inline void epilogue(v8f c, float biv, float bmv,
                                int mTile, int col, int lane,
                                unsigned short* __restrict__ Hout,      // bf16 [B,H]
                                float* __restrict__ Yf, size_t yStride, // nullable
                                float* __restrict__ Yf2) {              // nullable, stride H
    const int hi = lane >> 4;
#pragma unroll
    for (int i = 0; i < 8; ++i) {
        const int row = mTile * 16 + i + 8 * hi;
        float z = c[i] + biv;
        float mag = fabsf(z) + bmv;
        float y = (mag > 0.0f) ? ((z < 0.0f) ? -mag : mag) : 0.0f;   // modrelu
        Hout[(size_t)row * H_ + col] = f2bf(y);
        if (Yf)  Yf[(size_t)row * yStride + col] = y;
        if (Yf2) Yf2[(size_t)row * H_ + col] = y;
    }
}

// Monotonic-counter grid barrier (all NBLOCKS co-resident).
__device__ inline void grid_barrier(unsigned* cnt, unsigned barId) {
    __threadfence();          // publish this block's writes (release)
    __syncthreads();
    if (threadIdx.x == 0) {
        atomicAdd(cnt, 1u);
        const unsigned target = barId * gridDim.x;
        while (*(volatile unsigned*)cnt < target) {
            __builtin_amdgcn_s_sleep(1);
        }
    }
    __syncthreads();
    __threadfence();          // see other blocks' writes (acquire)
}

__global__ void __launch_bounds__(NTHREADS)
rnn_persistent(const float* __restrict__ x,
               const unsigned short* __restrict__ Wi0,
               const unsigned short* __restrict__ Wr0,
               const float* __restrict__ bi0, const float* __restrict__ bm0,
               const unsigned short* __restrict__ Wi1,
               const unsigned short* __restrict__ Wr1,
               const float* __restrict__ bi1, const float* __restrict__ bm1,
               unsigned short* __restrict__ xbf,    // 2 x [B,D] bf16 ping-pong
               unsigned short* __restrict__ h0buf,  // 2 x [B,H] bf16 ping-pong
               unsigned short* __restrict__ h1buf,  // 2 x [B,H] bf16 ping-pong
               float* __restrict__ out,             // [B,T,H] ++ [2,B,H]
               unsigned* __restrict__ barCnt) {
    // 256 KB weight slice in LDS: [Wi0 | Wr0 | Wi1 | Wr1], each [32 cols][1024 K]
    __shared__ unsigned short sW[4 * SLICE_];

    const int lane = threadIdx.x & 31;
    const int wave = threadIdx.x >> 5;
    const int job  = blockIdx.x * (NTHREADS / 32) + wave;  // 0..127
    const int mTile = job & 3;                              // 4 M-tiles (B=64)
    const int nBase = (job >> 2) * 32;                      // == blockIdx.x * 32
    const int gtid  = threadIdx.x + blockIdx.x * NTHREADS;  // 0..4095
    float* outTail = out + (size_t)B_ * T_ * H_;            // final states [2,B,H]

    // ---- preamble: stage this block's weight slices into LDS (once).
    {
        const unsigned short* wsrc[4] = {Wi0, Wr0, Wi1, Wr1};
#pragma unroll
        for (int wgt = 0; wgt < 4; ++wgt) {
            const unsigned short* src = wsrc[wgt] + (size_t)nBase * 1024;
            unsigned short* dst = sW + wgt * SLICE_;
            for (int q = threadIdx.x; q < SLICE_ / 8; q += NTHREADS) {
                *(uint4*)(dst + q * 8) = *(const uint4*)(src + q * 8);
            }
        }
        __syncthreads();
    }

    // ---- preload per-wave bias/modrelu params (columns are fixed per wave).
    const int col0 = nBase + (lane & 15);
    const int col1 = col0 + 16;
    const float bi0v0 = bi0[col0], bm0v0 = bm0[col0];
    const float bi0v1 = bi0[col1], bm0v1 = bm0[col1];
    const float bi1v0 = bi1[col0], bm1v0 = bm1[col0];
    const float bi1v1 = bi1[col1], bm1v1 = bm1[col1];

    unsigned barId = 0;
    for (int t = 0; t < T_; ++t) {
        const int r = t & 1, w = r ^ 1;
        const bool last = (t == T_ - 1);

        // ---- layer 0: h0 = modrelu(x_t @ Wi0^T + h0_prev @ Wr0^T + bi0, bm0)
        v8f c0 = {}; v8f c1 = {};
        gemm_tile(c0, c1, xbf + (size_t)r * BH_, sW + 0 * SLICE_, mTile, lane);
        gemm_tile(c0, c1, h0buf + (size_t)r * BH_, sW + 1 * SLICE_, mTile, lane);
        epilogue(c0, bi0v0, bm0v0, mTile, col0, lane, h0buf + (size_t)w * BH_,
                 nullptr, 0, last ? outTail : nullptr);
        epilogue(c1, bi0v1, bm0v1, mTile, col1, lane, h0buf + (size_t)w * BH_,
                 nullptr, 0, last ? outTail : nullptr);

        // ---- cooperative stage of x_{t+1} -> bf16 (read next step, 2 barriers away)
        if (!last) {
            const float* xs = x + (size_t)(t + 1) * D_;          // x[b, t+1, d]
            unsigned short* xd = xbf + (size_t)w * BH_;
            for (int i4 = gtid; i4 < BH_ / 4; i4 += NBLOCKS * NTHREADS) {
                const int idx = i4 * 4;
                const int b = idx >> 10, d = idx & 1023;
                float4 f = *(const float4*)(xs + (size_t)b * (T_ * D_) + d);
                uint2 pk;
                pk.x = (unsigned)f2bf(f.x) | ((unsigned)f2bf(f.y) << 16);
                pk.y = (unsigned)f2bf(f.z) | ((unsigned)f2bf(f.w) << 16);
                *(uint2*)(xd + idx) = pk;
            }
        }
        grid_barrier(barCnt, ++barId);

        // ---- layer 1: y_t = h1 = modrelu(h0 @ Wi1^T + h1_prev @ Wr1^T + bi1, bm1)
        c0 = (v8f){}; c1 = (v8f){};
        gemm_tile(c0, c1, h0buf + (size_t)w * BH_, sW + 2 * SLICE_, mTile, lane);
        gemm_tile(c0, c1, h1buf + (size_t)r * BH_, sW + 3 * SLICE_, mTile, lane);
        epilogue(c0, bi1v0, bm1v0, mTile, col0, lane, h1buf + (size_t)w * BH_,
                 out + (size_t)t * H_, (size_t)T_ * H_, last ? (outTail + BH_) : nullptr);
        epilogue(c1, bi1v1, bm1v1, mTile, col1, lane, h1buf + (size_t)w * BH_,
                 out + (size_t)t * H_, (size_t)T_ * H_, last ? (outTail + BH_) : nullptr);
        grid_barrier(barCnt, ++barId);
    }
}

__global__ void convert_bf16(const float* __restrict__ src,
                             unsigned short* __restrict__ dst, int n) {
    int i = blockIdx.x * blockDim.x + threadIdx.x;
    if (i < n) dst[i] = f2bf(src[i]);
}

// Zero bf16 state buffers (slot 0), stage x_0 into xbf slot 0, reset barrier.
__global__ void init_state(const float* __restrict__ x,
                           unsigned short* __restrict__ xbf,
                           unsigned short* __restrict__ h0,
                           unsigned short* __restrict__ h1,
                           unsigned* __restrict__ cnt) {
    int i = blockIdx.x * blockDim.x + threadIdx.x;
    if (i < BH_) {
        h0[i] = 0;
        h1[i] = 0;
        const int b = i >> 10, d = i & 1023;
        xbf[i] = f2bf(x[(size_t)b * (T_ * D_) + d]);   // x[b, 0, d]
    }
    if (i == 0) *cnt = 0;
}

extern "C" void kernel_launch(void* const* d_in, const int* in_sizes, int n_in,
                              void* d_out, int out_size, void* d_ws, size_t ws_size,
                              hipStream_t stream) {
    const float* x   = (const float*)d_in[0];
    const float* Wi0 = (const float*)d_in[1];
    const float* bi0 = (const float*)d_in[2];
    const float* Wr0 = (const float*)d_in[3];
    const float* bm0 = (const float*)d_in[4];
    const float* Wi1 = (const float*)d_in[5];
    const float* bi1 = (const float*)d_in[6];
    const float* Wr1 = (const float*)d_in[7];
    const float* bm1 = (const float*)d_in[8];
    float* out = (float*)d_out;

    const int WN = H_ * D_;        // 1M elements per weight
    // ws: 4 bf16 weights (8 MB) + bf16 ping-pong xbf/h0/h1 (3 x 256 KB) + counter
    unsigned short* Wi0c = (unsigned short*)d_ws;
    unsigned short* Wr0c = Wi0c + WN;
    unsigned short* Wi1c = Wr0c + WN;
    unsigned short* Wr1c = Wi1c + WN;
    unsigned short* xbf   = Wr1c + WN;        // 2 * B*H
    unsigned short* h0buf = xbf + 2 * BH_;    // 2 * B*H
    unsigned short* h1buf = h0buf + 2 * BH_;  // 2 * B*H
    unsigned* barCnt = (unsigned*)(h1buf + 2 * BH_);

    const int cb = 256, cg = (WN + cb - 1) / cb;
    convert_bf16<<<cg, cb, 0, stream>>>(Wi0, Wi0c, WN);
    convert_bf16<<<cg, cb, 0, stream>>>(Wr0, Wr0c, WN);
    convert_bf16<<<cg, cb, 0, stream>>>(Wi1, Wi1c, WN);
    convert_bf16<<<cg, cb, 0, stream>>>(Wr1, Wr1c, WN);
    init_state<<<(BH_ + cb - 1) / cb, cb, 0, stream>>>(x, xbf, h0buf, h1buf, barCnt);

    rnn_persistent<<<NBLOCKS, NTHREADS, 0, stream>>>(
        x, Wi0c, Wr0c, bi0, bm0, Wi1c, Wr1c, bi1, bm1,
        xbf, h0buf, h1buf, out, barCnt);
}
